// MultiHeadAttention_45131516346838
// MI455X (gfx1250) — matrix operand order repaired and checked by reading, not compiled, not run on hardware
//
#include <hip/hip_runtime.h>

// ---------------------------------------------------------------------------
// Graph-masked multi-head attention for MI455X (gfx1250, wave32, WMMA).
// B=8, N=1024, D=768, H=12, HD=64.
//
// Kernel 1: QKV projection GEMM (fp32 in, bf16 out), WMMA f32_16x16x32_bf16.
//           Q,K stored [b,h,n,hd]; V stored transposed [b,h,hd,n].
// Kernel 2: flash-style fused attention with double-buffered
//           global_load_async_to_lds_b128 staging of K/V chunks,
//           scores (WMMA) + beta*adj + mask + online softmax + P@V (WMMA).
// ---------------------------------------------------------------------------

typedef __attribute__((ext_vector_type(16))) __bf16 v16bf;
typedef __attribute__((ext_vector_type(8)))  __bf16 v8bf;
typedef __attribute__((ext_vector_type(8)))  float  v8f;

#define BATCH  8
#define NSEQ   1024
#define DMODEL 768
#define NHEAD  12
#define HDIM   64

// float -> bf16, round to nearest even
__device__ __forceinline__ unsigned short f2bf(float f) {
  union { float f; unsigned int u; } x; x.f = f;
  unsigned int u = x.u;
  return (unsigned short)((u + 0x7FFFu + ((u >> 16) & 1u)) >> 16);
}

// build a 16-element bf16 fragment from two 8-element (16B) runs
__device__ __forceinline__ v16bf ld16(const unsigned short* p0,
                                      const unsigned short* p1) {
  v8bf lo = *(const v8bf*)p0;
  v8bf hi = *(const v8bf*)p1;
  v16bf r;
#pragma unroll
  for (int i = 0; i < 8; ++i) { r[i] = lo[i]; r[i + 8] = hi[i]; }
  return r;
}

// CDNA5 async copy: 16B global -> LDS, bypassing VGPRs (tracked by ASYNCcnt).
// lds_byte_off is the low-32-bit LDS address of a generic shared pointer.
__device__ __forceinline__ void cp_async16(unsigned lds_byte_off, const void* gptr) {
  asm volatile("global_load_async_to_lds_b128 %0, %1, off"
               :: "v"(lds_byte_off), "v"(gptr)
               : "memory");
}
__device__ __forceinline__ void wait_async0() {
  asm volatile("s_wait_asynccnt 0x0" ::: "memory");
}

// ---------------------------------------------------------------------------
// Kernel 1: Y = X @ W + b, output bf16.
// grid = (8192/64, 768/64, 3), block = 128 (4 waves).
// ---------------------------------------------------------------------------
__global__ __launch_bounds__(128)
void qkv_proj_kernel(const float* __restrict__ X,
                     const float* __restrict__ Wq, const float* __restrict__ bq,
                     const float* __restrict__ Wk, const float* __restrict__ bk,
                     const float* __restrict__ Wv, const float* __restrict__ bv,
                     unsigned short* __restrict__ ws) {
  __shared__ unsigned short Ash[64 * 32];  // [row][k]  (bf16)
  __shared__ unsigned short Bsh[64 * 32];  // [col][k]  (bf16, transposed W)

  const int t    = threadIdx.x;
  const int wave = t >> 5;
  const int lane = t & 31;
  const int g    = lane >> 4;
  const int ln   = lane & 15;
  const int rowBase = blockIdx.x * 64;
  const int colBase = blockIdx.y * 64;
  const int wsel    = blockIdx.z;

  const float* W    = (wsel == 0) ? Wq : ((wsel == 1) ? Wk : Wv);
  const float* bias = (wsel == 0) ? bq : ((wsel == 1) ? bk : bv);

  v8f c[4] = {};

  for (int kk = 0; kk < DMODEL; kk += 32) {
    __syncthreads();
    // stage X tile 64x32 (fp32 -> bf16), row-major
    for (int i = t; i < 64 * 32; i += 128) {
      int r = i >> 5, cc = i & 31;
      Ash[i] = f2bf(X[(size_t)(rowBase + r) * DMODEL + kk + cc]);
    }
    // stage W tile 32x64 transposed -> [col][k]
    for (int i = t; i < 64 * 32; i += 128) {
      int k = i >> 6, cc = i & 63;
      Bsh[cc * 32 + k] = f2bf(W[(size_t)(kk + k) * DMODEL + colBase + cc]);
    }
    __syncthreads();

    const int arow = wave * 16 + ln;
    // A fragment: element i -> K = 16*(i>>3) + 8*g + (i&7)
    v16bf a = ld16(&Ash[arow * 32 + 8 * g], &Ash[arow * 32 + 16 + 8 * g]);
#pragma unroll
    for (int ct = 0; ct < 4; ++ct) {
      int col = ct * 16 + ln;
      // B fragment: element i -> K = 16*(i>>3) + 8*g + (i&7)
      v16bf b = ld16(&Bsh[col * 32 + 8 * g], &Bsh[col * 32 + 16 + 8 * g]);
      c[ct] = __builtin_amdgcn_wmma_f32_16x16x32_bf16(
          false, a, false, b, (short)0, c[ct], false, false);
    }
  }

  unsigned short* Qo = ws;
  unsigned short* Ko = ws + (size_t)BATCH * NHEAD * NSEQ * HDIM;
  unsigned short* Vo = ws + (size_t)2 * BATCH * NHEAD * NSEQ * HDIM;

#pragma unroll
  for (int ct = 0; ct < 4; ++ct) {
    int gc = colBase + ct * 16 + ln;
    int h = gc >> 6, hd = gc & 63;
    float bb = bias[gc];
#pragma unroll
    for (int v = 0; v < 8; ++v) {
      int gr = rowBase + wave * 16 + v + 8 * g;
      int b = gr >> 10, n = gr & 1023;
      unsigned short val = f2bf(c[ct][v] + bb);
      if (wsel == 2)       // V transposed: [b,h,hd,n]
        Vo[(((size_t)(b * NHEAD + h)) * HDIM + hd) * NSEQ + n] = val;
      else if (wsel == 1)  // K: [b,h,n,hd]
        Ko[(((size_t)(b * NHEAD + h)) * NSEQ + n) * HDIM + hd] = val;
      else                 // Q: [b,h,n,hd]
        Qo[(((size_t)(b * NHEAD + h)) * NSEQ + n) * HDIM + hd] = val;
    }
  }
}

// ---------------------------------------------------------------------------
// Kernel 2: fused attention, flash-style over 64-key chunks with
// double-buffered async K/V staging.
// grid = B*H*(N/128) = 768 blocks, block = 256 (8 waves).
// ---------------------------------------------------------------------------
__global__ __launch_bounds__(256)
void attn_kernel(const float* __restrict__ adj,
                 const float* __restrict__ betap,
                 const unsigned short* __restrict__ ws,
                 float* __restrict__ out) {
  __shared__ unsigned short Ksh[2][64 * 64];     // [buf][key][hd]
  __shared__ unsigned short Vsh[2][64 * 64];     // [buf][hd][key]
  __shared__ unsigned short Psh[8 * 16 * 64];    // per-wave P tile [row][key]

  const int t    = threadIdx.x;
  const int wave = t >> 5;
  const int lane = t & 31;
  const int g    = lane >> 4;
  const int ln   = lane & 15;

  const int tile = blockIdx.x;
  const int qblk = tile & 7;             // N/128 = 8 tiles
  const int h    = (tile >> 3) % NHEAD;
  const int b    = tile / (8 * NHEAD);

  const float beta = betap[0];
  const size_t SZ = (size_t)BATCH * NHEAD * NSEQ * HDIM;
  const size_t bh = (size_t)(b * NHEAD + h);
  const unsigned short* Qb = ws + bh * NSEQ * HDIM;
  const unsigned short* Kb = ws + SZ + bh * NSEQ * HDIM;
  const unsigned short* Vb = ws + 2 * SZ + bh * HDIM * NSEQ;

  const int qbase = qblk * 128 + wave * 16;

  // issue async copies of one 64-key K/V chunk into LDS buffer `buf`
  auto issue_chunk = [&](int buf, int keybase) {
    const unsigned kl = (unsigned)(uintptr_t)(&Ksh[buf][0]);
    const unsigned vl = (unsigned)(uintptr_t)(&Vsh[buf][0]);
    const uint4* ksrc = (const uint4*)(Kb + (size_t)keybase * HDIM);
#pragma unroll
    for (int i = t; i < 512; i += 256)
      cp_async16(kl + (unsigned)i * 16u, (const void*)(ksrc + i));
#pragma unroll
    for (int i = t; i < 512; i += 256) {
      int r = i >> 3, s8 = i & 7;
      cp_async16(vl + (unsigned)i * 16u,
                 (const void*)((const uint4*)(Vb + (size_t)r * NSEQ + keybase) + s8));
    }
  };

  // Q fragments (rows qbase..qbase+15, hd split in two K=32 fragments)
  v16bf aq[2];
#pragma unroll
  for (int f = 0; f < 2; ++f) {
    const unsigned short* qp = Qb + (size_t)(qbase + ln) * HDIM + 32 * f + 8 * g;
    aq[f] = ld16(qp, qp + 16);
  }

  float m[8], l[8];
  v8f acc[4] = {};
#pragma unroll
  for (int v = 0; v < 8; ++v) { m[v] = -3.0e38f; l[v] = 0.0f; }

  issue_chunk(0, 0);  // prime the pipeline

  for (int kc = 0; kc < NSEQ / 64; ++kc) {
    const int cur = kc & 1;
    const int keybase = kc * 64;

    wait_async0();      // this wave's chunk-kc async copies landed in LDS
    __syncthreads();    // all waves' copies visible; prev-chunk reads done
    if (kc + 1 < NSEQ / 64) issue_chunk(cur ^ 1, keybase + 64);  // overlap

    const unsigned short* Kc = &Ksh[cur][0];
    const unsigned short* Vc = &Vsh[cur][0];

    // scores: 4 key sub-tiles x (HD=64 -> 2 WMMAs of K=32)
    v8f s[4] = {};
#pragma unroll
    for (int st = 0; st < 4; ++st) {
#pragma unroll
      for (int f = 0; f < 2; ++f) {
        const unsigned short* kp = &Kc[(st * 16 + ln) * 64 + 32 * f + 8 * g];
        v16bf bk = ld16(kp, kp + 16);
        s[st] = __builtin_amdgcn_wmma_f32_16x16x32_bf16(
            false, aq[f], false, bk, (short)0, s[st], false, false);
      }
    }

    // scale + beta*adj + adjacency mask (+ prefetch next chunk's adj rows)
#pragma unroll
    for (int st = 0; st < 4; ++st) {
      int key = keybase + st * 16 + ln;
#pragma unroll
      for (int v = 0; v < 8; ++v) {
        int row = qbase + v + 8 * g;
        const float* aptr = &adj[((size_t)b * NSEQ + row) * NSEQ + key];
        if (st == 0 && kc + 1 < NSEQ / 64)
          __builtin_prefetch(&adj[((size_t)b * NSEQ + row) * NSEQ +
                                  keybase + 64 + lane * 2], 0, 1);
        float aval = *aptr;
        float sc = s[st][v] * 0.125f + beta * aval;  // 1/sqrt(64) = 0.125
        s[st][v] = (aval > 0.0f) ? sc : -9.0e15f;
      }
    }

    // online softmax per row (rows split: lanes 0-15 row v, 16-31 row v+8)
    float alpha[8];
#pragma unroll
    for (int v = 0; v < 8; ++v) {
      float mr = fmaxf(fmaxf(s[0][v], s[1][v]), fmaxf(s[2][v], s[3][v]));
#pragma unroll
      for (int off = 8; off >= 1; off >>= 1)
        mr = fmaxf(mr, __shfl_xor(mr, off, 32));
      float mn = fmaxf(m[v], mr);
      alpha[v] = __expf(m[v] - mn);
      m[v] = mn;
      float sr = 0.0f;
#pragma unroll
      for (int st = 0; st < 4; ++st) {
        float p = __expf(s[st][v] - mn);
        s[st][v] = p;
        sr += p;
      }
#pragma unroll
      for (int off = 8; off >= 1; off >>= 1)
        sr += __shfl_xor(sr, off, 32);
      l[v] = l[v] * alpha[v] + sr;
    }
#pragma unroll
    for (int tb = 0; tb < 4; ++tb)
#pragma unroll
      for (int v = 0; v < 8; ++v) acc[tb][v] *= alpha[v];

    // write P (bf16) to per-wave LDS scratch in row-major [row][key]
    unsigned short* Pw = &Psh[wave * 1024];
#pragma unroll
    for (int st = 0; st < 4; ++st)
#pragma unroll
      for (int v = 0; v < 8; ++v)
        Pw[(v + 8 * g) * 64 + st * 16 + ln] = f2bf(s[st][v]);

    // acc += P(16x64) @ V(64x64): 2 K=32 steps x 4 hd tiles
#pragma unroll
    for (int f = 0; f < 2; ++f) {
      const unsigned short* pp = Pw + ln * 64 + 32 * f + 8 * g;
      v16bf ap = ld16(pp, pp + 16);
#pragma unroll
      for (int tb = 0; tb < 4; ++tb) {
        const unsigned short* vp = &Vc[(tb * 16 + ln) * 64 + 32 * f + 8 * g];
        v16bf bv = ld16(vp, vp + 16);
        acc[tb] = __builtin_amdgcn_wmma_f32_16x16x32_bf16(
            false, ap, false, bv, (short)0, acc[tb], false, false);
      }
    }
  }

  // epilogue: normalize and write fp32 output [b, n, h*64+hd]
#pragma unroll
  for (int v = 0; v < 8; ++v) l[v] = 1.0f / l[v];
#pragma unroll
  for (int tb = 0; tb < 4; ++tb) {
    int col = h * 64 + tb * 16 + ln;
#pragma unroll
    for (int v = 0; v < 8; ++v) {
      int row = qbase + v + 8 * g;
      out[((size_t)b * NSEQ + row) * DMODEL + col] = acc[tb][v] * l[v];
    }
  }
}

// ---------------------------------------------------------------------------
extern "C" void kernel_launch(void* const* d_in, const int* in_sizes, int n_in,
                              void* d_out, int out_size, void* d_ws, size_t ws_size,
                              hipStream_t stream) {
  const float* X    = (const float*)d_in[0];
  const float* adj  = (const float*)d_in[1];
  const float* Wq   = (const float*)d_in[2];
  const float* bq   = (const float*)d_in[3];
  const float* Wk   = (const float*)d_in[4];
  const float* bk   = (const float*)d_in[5];
  const float* Wv   = (const float*)d_in[6];
  const float* bv   = (const float*)d_in[7];
  const float* beta = (const float*)d_in[8];
  unsigned short* ws = (unsigned short*)d_ws;
  float* out = (float*)d_out;

  // QKV projections: (8192x768)@(768x768) x3, bf16 intermediates in ws
  dim3 g1(8192 / 64, DMODEL / 64, 3);
  qkv_proj_kernel<<<g1, 128, 0, stream>>>(X, Wq, bq, Wk, bk, Wv, bv, ws);

  // Fused attention: 8*12*8 = 768 blocks of 8 waves
  attn_kernel<<<BATCH * NHEAD * (NSEQ / 128), 256, 0, stream>>>(adj, beta, ws, out);
}
